// SNAT3_80857054314860
// MI455X (gfx1250) — matrix operand order repaired and checked by reading, not compiled, not run on hardware
//
#include <hip/hip_runtime.h>
#include <math.h>
#include <stdint.h>

// ---------------- problem constants (match reference) ----------------
#define IN_DIM 128
#define HID    64
#define SLOPE  0.2f
#define THR    0.01f
#define NCONV  4               // NUM_LAYERS + 1
#define DCAT   (NCONV * HID)   // 256

typedef __attribute__((ext_vector_type(16))) _Float16 v16h;
typedef __attribute__((ext_vector_type(8)))  float    v8f;
typedef __attribute__((ext_vector_type(4)))  int      v4i;

// pointer-to-AS1 / pointer-to-AS3 int4 (for async global->LDS builtin)
typedef __attribute__((address_space(1))) v4i* gv4i_p;
typedef __attribute__((address_space(3))) v4i* lv4i_p;

// ---- optional CDNA5 async global->LDS path (ASYNCcnt) ----
#if defined(__has_builtin)
#if __has_builtin(__builtin_amdgcn_global_load_async_to_lds_b128) && \
    __has_builtin(__builtin_amdgcn_s_wait_asynccnt)
#define USE_ASYNC_LDS 1
#endif
#endif

__device__ __forceinline__ void g2lds_16B(const float* g, float* l)
{
#ifdef USE_ASYNC_LDS
    __builtin_amdgcn_global_load_async_to_lds_b128(
        (gv4i_p)(uintptr_t)g,
        (lv4i_p)(uintptr_t)(uint32_t)(uintptr_t)l,   // LDS offset = low 32 bits
        0, 0);
#else
    l[0] = g[0]; l[1] = g[1]; l[2] = g[2]; l[3] = g[3];
#endif
}

__device__ __forceinline__ void lds_stage_fence()
{
#ifdef USE_ASYNC_LDS
    __builtin_amdgcn_s_wait_asynccnt(0);
#endif
    __syncthreads();
}

// ---------------------------------------------------------------------
// WMMA GEMM:  C[M x 64] = ACT( A[M x K] @ B[K x 64] + bias )
// A row-major (leading dim lda), B row-major ldb=64, C ldc=64.
// blockDim = 128 (4 waves); wave w computes 16x16 tile at col 16*w.
// A stripe staged f32 in LDS (async b128 when available);
// B panel staged f16 transposed in LDS; K-loop fully unrolled -> K/32 WMMAs.
// ACT: 0=none, 1=tanh, 2=relu.
// ---------------------------------------------------------------------
template<int K, int ACT>
__global__ void __launch_bounds__(128)
gemm_wmma(const float* __restrict__ A, const float* __restrict__ B,
          const float* __restrict__ bias, float* __restrict__ C,
          int M, int lda)
{
    constexpr int APAD = 4;   // floats  -> 16-lane bank spread on A reads
    constexpr int BPAD = 8;   // halfs   -> keeps pairs aligned, spreads banks
    __shared__ float    As[16 * (K + APAD)];
    __shared__ _Float16 Bs[64 * (K + BPAD)];

    const int tid  = threadIdx.x;
    const int wave = tid >> 5;            // 0..3 -> column tile
    const int lane = tid & 31;
    const int row0 = blockIdx.x * 16;
    const int n0   = wave * 16;
    const int m    = lane & 15;
    const int hi   = lane >> 4;           // lane half

    // ---- stage A stripe: 16 rows x K f32, 16B chunks, async when available
    constexpr int CHUNKS = 16 * K / 4;    // 16-byte chunks
#pragma unroll
    for (int it = 0; it < CHUNKS / 128; ++it) {
        int c   = tid + 128 * it;
        int r   = c / (K / 4);
        int c16 = c % (K / 4);
        int gr  = row0 + r; if (gr >= M) gr = M - 1;
        g2lds_16B(A + (size_t)gr * lda + c16 * 4, &As[r * (K + APAD) + c16 * 4]);
    }
    // ---- stage B panel: f16, transposed [n][k] so lane K-runs are contiguous
#pragma unroll 4
    for (int idx = tid; idx < 64 * K; idx += 128) {
        int k = idx >> 6, n = idx & 63;
        Bs[n * (K + BPAD) + k] = (_Float16)B[idx];
    }
    lds_stage_fence();

    v8f c = {};
#pragma unroll
    for (int k0 = 0; k0 < K; k0 += 32) {
        // A fragment 16x32: lanes0-15 K=0..7/16..23 ; lanes16-31 K=8..15/24..31
        v16h a;
        const float* Ar = &As[m * (K + APAD) + k0 + hi * 8];
#pragma unroll
        for (int i = 0; i < 4; ++i) {
            a[2*i+0]   = (_Float16)Ar[2*i+0];
            a[2*i+1]   = (_Float16)Ar[2*i+1];
            a[8+2*i+0] = (_Float16)Ar[16 + 2*i+0];
            a[8+2*i+1] = (_Float16)Ar[16 + 2*i+1];
        }
        // B fragment 32x16: lanes0-15 K=0..7/16..23 ; lanes16-31 K=8..15/24..31
        v16h b;
        const _Float16* Br = &Bs[(n0 + m) * (K + BPAD) + k0 + hi * 8];
#pragma unroll
        for (int j = 0; j < 4; ++j) { b[2*j] = Br[2*j]; b[2*j+1] = Br[2*j+1]; b[8+2*j] = Br[16+2*j]; b[8+2*j+1] = Br[16+2*j+1]; }

        c = __builtin_amdgcn_wmma_f32_16x16x32_f16(false, a, false, b,
                                                   (short)0, c, false, false);
    }

    // ---- epilogue: lane n = m ; row = vgpr + 8*hi
    const float bv = bias ? bias[n0 + m] : 0.0f;
#pragma unroll
    for (int v = 0; v < 8; ++v) {
        int mm = row0 + v + hi * 8;
        if (mm < M) {
            float val = c[v] + bv;
            if (ACT == 1)      val = tanhf(val);
            else if (ACT == 2) val = fmaxf(val, 0.0f);
            C[(size_t)mm * 64 + n0 + m] = val;
        }
    }
}

// ---------------------------------------------------------------------
__device__ __forceinline__ void atomicMaxF(float* addr, float value)
{
    if (value >= 0.0f)
        atomicMax((int*)addr, __float_as_int(value));
    else
        atomicMin((unsigned int*)addr, (unsigned int)__float_as_int(value));
}

// el/er dot products per node + per-layer segment init
__global__ void __launch_bounds__(256)
node_dots(const float* __restrict__ feat, const float* __restrict__ al,
          const float* __restrict__ ar, float* __restrict__ el,
          float* __restrict__ er, float* __restrict__ segmax,
          float* __restrict__ segsum, int Nn)
{
    int n = blockIdx.x * blockDim.x + threadIdx.x;
    if (n >= Nn) return;
    const float* f = feat + (size_t)n * 64;
    float sl = 0.f, sr = 0.f;
#pragma unroll 8
    for (int d = 0; d < 64; ++d) { float v = f[d]; sl += v * al[d]; sr += v * ar[d]; }
    el[n] = sl; er[n] = sr;
    segmax[n] = -1e30f;
    segsum[n] = 0.0f;
}

__global__ void __launch_bounds__(256)
zero_buf(float* __restrict__ p, int count)
{
    int i = blockIdx.x * blockDim.x + threadIdx.x;
    if (i < count) p[i] = 0.0f;
}

// edge raw score: leaky_relu(el[src]+er[dst]), optional prune mask, segmax(dst)
__global__ void __launch_bounds__(256)
edge_score(const int* __restrict__ src, const int* __restrict__ dst,
           const float* __restrict__ el, const float* __restrict__ er,
           const float* __restrict__ mask, float* __restrict__ e,
           float* __restrict__ segmax, int Ne)
{
    int i = blockIdx.x * blockDim.x + threadIdx.x;
    if (i >= Ne) return;
    float v = el[src[i]] + er[dst[i]];
    v = (v > 0.f) ? v : SLOPE * v;
    if (mask && mask[i] == 0.0f) v = -1e9f;
    e[i] = v;
    atomicMaxF(&segmax[dst[i]], v);
}

// alpha_unnorm = exp(e - segmax[dst]); segsum(dst) += alpha
__global__ void __launch_bounds__(256)
edge_exp(const int* __restrict__ dst, const float* __restrict__ e,
         const float* __restrict__ segmax, float* __restrict__ alpha,
         float* __restrict__ segsum, int Ne)
{
    int i = blockIdx.x * blockDim.x + threadIdx.x;
    if (i >= Ne) return;
    int dn = dst[i];
    float a = __expf(e[i] - segmax[dn]);
    alpha[i] = a;
    atomicAdd(&segsum[dn], a);
}

// scatter: acc[dst] += feat[src] * alpha_norm ; layer0 also emits prune mask
__global__ void __launch_bounds__(256)
edge_scatter(const int* __restrict__ src, const int* __restrict__ dst,
             const float* __restrict__ feat, const float* __restrict__ alpha,
             const float* __restrict__ segsum, float* __restrict__ mask_out,
             float* __restrict__ acc, int Ne)
{
    int gid = blockIdx.x * blockDim.x + threadIdx.x;
    int i = gid >> 6;            // edge
    int d = gid & 63;            // feature dim
    if (i >= Ne) return;
    int dn = dst[i];
    float a = alpha[i] / fmaxf(segsum[dn], 1e-9f);
    if (mask_out && d == 0) mask_out[i] = (a >= THR) ? 1.0f : 0.0f;
    atomicAdd(&acc[(size_t)dn * 64 + d], feat[(size_t)src[i] * 64 + d] * a);
}

// elu and write into concat buffer column block (row stride DCAT)
__global__ void __launch_bounds__(256)
elu_store(const float* __restrict__ acc, float* __restrict__ hcat_col, int Nn)
{
    int gid = blockIdx.x * blockDim.x + threadIdx.x;
    int n = gid >> 6, d = gid & 63;
    if (n >= Nn) return;
    float v = acc[(size_t)n * 64 + d];
    v = (v > 0.f) ? v : (__expf(v) - 1.0f);
    hcat_col[(size_t)n * DCAT + d] = v;
}

// final 64 -> 1 projection + relu
__global__ void __launch_bounds__(256)
final_proj(const float* __restrict__ in, const float* __restrict__ W2,
           const float* __restrict__ b2, float* __restrict__ out, int Nn)
{
    int n = blockIdx.x * blockDim.x + threadIdx.x;
    if (n >= Nn) return;
    const float* r = in + (size_t)n * 64;
    float s = b2[0];
#pragma unroll 8
    for (int d = 0; d < 64; ++d) s += r[d] * W2[d];
    out[n] = fmaxf(s, 0.0f);
}

// ---------------------------------------------------------------------
extern "C" void kernel_launch(void* const* d_in, const int* in_sizes, int n_in,
                              void* d_out, int out_size, void* d_ws, size_t ws_size,
                              hipStream_t stream)
{
    const float* x       = (const float*)d_in[0];
    const int*   e_src   = (const int*)  d_in[1];
    const int*   e_dst   = (const int*)  d_in[2];
    const float* W_embed = (const float*)d_in[3];
    const float* W_gat   = (const float*)d_in[4];   // [4,64,64]
    const float* a_l     = (const float*)d_in[5];   // [4,1,64]
    const float* a_r     = (const float*)d_in[6];
    const float* W0      = (const float*)d_in[7];   // [256,64]
    const float* b0      = (const float*)d_in[8];
    const float* W1      = (const float*)d_in[9];   // [64,64]
    const float* b1      = (const float*)d_in[10];
    const float* W2      = (const float*)d_in[11];  // [64,1]
    const float* b2      = (const float*)d_in[12];
    float* out = (float*)d_out;

    const int N = in_sizes[0] / IN_DIM;   // 50000
    const int E = in_sizes[1];            // 800000

    // ---- workspace layout (floats) ----
    float* ws = (float*)d_ws;
    size_t off = 0;
    float* h_embed = ws + off; off += (size_t)N * 64;     // tanh(x@W_embed)
    float* hcat    = ws + off; off += (size_t)N * DCAT;   // [h0|h1|h2|feats]
    float* feat    = ws + off; off += (size_t)N * 64;     // h@W_gat / head tmp
    float* acc     = ws + off; off += (size_t)N * 64;     // segment-sum accum
    float* el      = ws + off; off += (size_t)N;
    float* er      = ws + off; off += (size_t)N;
    float* segmax  = ws + off; off += (size_t)N;
    float* segsum  = ws + off; off += (size_t)N;
    float* e_buf   = ws + off; off += (size_t)E;
    float* alpha   = ws + off; off += (size_t)E;
    float* mask    = ws + off; off += (size_t)E;
    (void)ws_size; (void)n_in; (void)out_size;

    const int gemmGrid = (N + 15) / 16;            // 3125 blocks x 128 thr
    const int nodeGrid = (N + 255) / 256;
    const int nvGrid   = (N * 64 + 255) / 256;
    const int edgeGrid = (E + 255) / 256;
    const int evGrid   = (E * 64 + 255) / 256;     // edge x dim threads

    // 1) embed: h = tanh(x @ W_embed)     K=128, act=tanh
    gemm_wmma<IN_DIM, 1><<<gemmGrid, 128, 0, stream>>>(x, W_embed, nullptr,
                                                       h_embed, N, IN_DIM);

    // 2) four SNAT conv layers
    for (int l = 0; l < NCONV; ++l) {
        const float* h_in  = (l == 0) ? h_embed : (hcat + (size_t)(l - 1) * 64);
        const int    lda   = (l == 0) ? 64 : DCAT;
        const float* msk_in  = (l == 0) ? nullptr : mask;
        float*       msk_out = (l == 0) ? mask : nullptr;

        // feat = h_in @ W_gat[l]   (WMMA, no activation)
        gemm_wmma<HID, 0><<<gemmGrid, 128, 0, stream>>>(h_in,
                                                        W_gat + (size_t)l * 64 * 64,
                                                        nullptr, feat, N, lda);
        // el/er + segment init
        node_dots<<<nodeGrid, 256, 0, stream>>>(feat, a_l + (size_t)l * 64,
                                                a_r + (size_t)l * 64,
                                                el, er, segmax, segsum, N);
        zero_buf<<<nvGrid, 256, 0, stream>>>(acc, N * 64);
        // edge softmax with pruning
        edge_score<<<edgeGrid, 256, 0, stream>>>(e_src, e_dst, el, er, msk_in,
                                                 e_buf, segmax, E);
        edge_exp<<<edgeGrid, 256, 0, stream>>>(e_dst, e_buf, segmax, alpha, segsum, E);
        edge_scatter<<<evGrid, 256, 0, stream>>>(e_src, e_dst, feat, alpha, segsum,
                                                 msk_out, acc, E);
        // elu -> concat column l
        elu_store<<<nvGrid, 256, 0, stream>>>(acc, hcat + (size_t)l * 64, N);
    }

    // 3) MLP head
    // out0 = relu(hcat @ W0 + b0)   K=256  -> feat
    gemm_wmma<DCAT, 2><<<gemmGrid, 128, 0, stream>>>(hcat, W0, b0, feat, N, DCAT);
    // out1 = relu(out0 @ W1 + b1)   K=64   -> acc
    gemm_wmma<HID, 2><<<gemmGrid, 128, 0, stream>>>(feat, W1, b1, acc, N, 64);
    // out  = relu(out1 @ W2 + b2)
    final_proj<<<nodeGrid, 256, 0, stream>>>(acc, W2, b2, out, N);
}
